// DeltaTokenShift_18356690223134
// MI455X (gfx1250) — matrix operand order repaired and checked by reading, not compiled, not run on hardware
//
#include <hip/hip_runtime.h>
#include <hip/hip_bf16.h>

typedef __attribute__((ext_vector_type(16))) __bf16 v16bf;
typedef __attribute__((ext_vector_type(8)))  float  v8f;
typedef __attribute__((ext_vector_type(4)))  unsigned int v4u;
typedef __attribute__((ext_vector_type(8)))  int  v8i;
typedef __attribute__((ext_vector_type(4)))  int  v4i;

#define BB 4
#define LL 4096
#define DD 1024
#define BL (BB * LL)            // 16384 tokens
#define BD (BB * DD)            // 4096 channels
#define NCHUNK 32
#define CHUNK (LL / NCHUNK)     // 128
#define PACK_PART (1u << 20)    // 1M bf16 elements per (kernel,part)

#if defined(__gfx1250__) && __has_builtin(__builtin_amdgcn_tensor_load_to_lds) && \
    __has_builtin(__builtin_amdgcn_s_wait_tensorcnt)
#define USE_TDM 1
#else
#define USE_TDM 0
#endif

// ---------------------------------------------------------------------------
// Kernel 0: repack erase/write weights into bf16 hi/lo, pre-swizzled into the
// per-lane B-operand layout kernel 1 uses (contiguous k, matching its A):
//   group(kc, col, half) = 16 contiguous bf16 = W[kc*32 + half*16 + i][col]
// so a lane's whole v16bf B operand is one contiguous 32-byte load.
// ---------------------------------------------------------------------------
__global__ void __launch_bounds__(256)
pack_w_kernel(const float* __restrict__ We, const float* __restrict__ Ww,
              __bf16* __restrict__ pack) {
    int t    = blockIdx.x * blockDim.x + threadIdx.x;   // 131072 threads
    int half = t & 1;
    int col  = (t >> 1) & (DD - 1);
    int kc   = (t >> 11) & 31;
    int kern = t >> 16;
    const float* W = kern ? Ww : We;
    int kbase = kc * 32 + half * 16;
    size_t gidx = (((size_t)kc * DD + col) * 2 + half) * 16;
    __bf16* hi = pack + (size_t)(kern * 2 + 0) * PACK_PART + gidx;
    __bf16* lo = pack + (size_t)(kern * 2 + 1) * PACK_PART + gidx;
#pragma unroll
    for (int i = 0; i < 16; ++i) {
        float f  = W[(size_t)(kbase + i) * DD + col];
        __bf16 h = (__bf16)f;
        hi[i] = h;
        lo[i] = (__bf16)(f - (float)h);
    }
}

// ---------------------------------------------------------------------------
// Kernel 1: fused dual-gate GEMM + sigmoid epilogue.
//   Block = 256 threads (8 waves). Block tile: M=16 tokens, N=512 cols.
//   Each wave owns four 16-col N-tiles x 2 gates (8 v8f accumulators).
//   A tile (16x32 f32) staged in LDS by the Tensor Data Mover (TDM descriptor,
//   pad_interval=32dw/pad_amount=1dw -> [16][33] layout), fallback = coop load.
//   2-term split: A*B ~= Ahi*Bhi + Ahi*Blo + Alo*Bhi  (3 WMMAs / product).
//   Writes a = sigmoid(-logitE) = 1-erase  and  c = write * x  to workspace.
// ---------------------------------------------------------------------------
__global__ void __launch_bounds__(256)
gates_gemm_kernel(const float* __restrict__ x,
                  const float* __restrict__ be, const float* __restrict__ bw,
                  const __bf16* __restrict__ pack,
                  float* __restrict__ a_buf, float* __restrict__ c_buf) {
    __shared__ float lds_x[16][33];   // +1 pad: matches TDM pad feature

    int tid    = threadIdx.x;
    int lane   = tid & 31;
    int wave   = tid >> 5;
    int lanelo = lane & 15;
    int laneHi = lane >> 4;
    int m0     = blockIdx.x * 16;                    // token block
    int colb   = blockIdx.y * 512 + wave * 64 + lanelo;

    v8f accE[4] = {v8f{}, v8f{}, v8f{}, v8f{}};
    v8f accW[4] = {v8f{}, v8f{}, v8f{}, v8f{}};

    const __bf16* pEh = pack + 0 * (size_t)PACK_PART;
    const __bf16* pEl = pack + 1 * (size_t)PACK_PART;
    const __bf16* pWh = pack + 2 * (size_t)PACK_PART;
    const __bf16* pWl = pack + 3 * (size_t)PACK_PART;

#if USE_TDM
    // LDS byte address of the staging tile (AS(3) pointers are 32-bit)
    unsigned lds_base =
        (unsigned)(size_t)(__attribute__((address_space(3))) void*)&lds_x[0][0];
#endif

    for (int kc = 0; kc < DD / 32; ++kc) {
#if USE_TDM
        if (wave == 0) {
            // --- Tensor DMA descriptor: 2D tile, 32 (K, contiguous) x 16 rows,
            //     f32 elements, row stride = DD, LDS pad 1 dword / 32 dwords.
            unsigned long long ga =
                (unsigned long long)(const void*)(x + (size_t)m0 * DD + kc * 32);
            v4u g0;
            g0[0] = 1u;                                   // count=1, user mode
            g0[1] = lds_base;                             // lds_addr
            g0[2] = (unsigned)(ga & 0xFFFFFFFFu);         // global_addr[31:0]
            g0[3] = (unsigned)((ga >> 32) & 0x01FFFFFFu)  // global_addr[56:32]
                    | 0x80000000u;                        // type=2 ("image")
            v8i g1;
            g1[0] = (int)(0x00020000u        // data_size=2 (4B), wg_mask=0
                    | (1u << 20)             // pad_enable
                    | (4u << 22)             // pad_interval: 32 dwords
                    | (0u << 25));           // pad_amount: 1 dword
            g1[1] = (int)((DD & 0xFFFFu) << 16);          // tensor_dim0 lo16<<16
            g1[2] = (int)(((DD >> 16) & 0xFFFFu)
                    | ((BL & 0xFFFFu) << 16));            // dim0 hi | dim1 lo
            g1[3] = (int)(((BL >> 16) & 0xFFFFu)
                    | (32u << 16));                       // dim1 hi | tile_dim0=32
            g1[4] = 16;                                   // tile_dim1=16 rows
            g1[5] = DD;                                   // tensor_dim0_stride
            g1[6] = 0;
            g1[7] = 0;
            v4i gz4 = {0, 0, 0, 0};                       // 2D: groups 2/3 unused
            v8i gz8 = {0, 0, 0, 0, 0, 0, 0, 0};           // aux group (6-arg form)
            __builtin_amdgcn_tensor_load_to_lds(g0, g1, gz4, gz4, gz8, 0);
            __builtin_amdgcn_s_wait_tensorcnt(0);
        }
#else
        // cooperative stage of the 16x32 f32 A tile
#pragma unroll
        for (int i = 0; i < 2; ++i) {
            int e = tid * 2 + i;
            int r = e >> 5, k = e & 31;
            lds_x[r][k] = x[(size_t)(m0 + r) * DD + kc * 32 + k];
        }
#endif
        __syncthreads();

        // build A hi/lo in this kernel's A layout (contiguous k, matches packed B):
        // elem j -> K = j + (laneHi ? 16 : 0)
        v16bf ahi, alo;
#pragma unroll
        for (int j = 0; j < 16; ++j) {
            int K   = j + laneHi * 16;
            float f = lds_x[lanelo][K];
            __bf16 h = (__bf16)f;
            ahi[j] = h;
            alo[j] = (__bf16)(f - (float)h);
        }
        __syncthreads();   // all reads done before next iteration's fill

        // B operands: lane<16 -> (col, K 0..15), lane>=16 -> (col, K 16..31)
#pragma unroll
        for (int t = 0; t < 4; ++t) {
            int col = colb + t * 16;
            size_t g = (((size_t)kc * DD + col) * 2 + laneHi) * 16;
            v16bf bh = *(const v16bf*)(pEh + g);
            v16bf bl = *(const v16bf*)(pEl + g);
            accE[t] = __builtin_amdgcn_wmma_f32_16x16x32_bf16(false, ahi, false, bh, (short)0, accE[t], false, false);
            accE[t] = __builtin_amdgcn_wmma_f32_16x16x32_bf16(false, ahi, false, bl, (short)0, accE[t], false, false);
            accE[t] = __builtin_amdgcn_wmma_f32_16x16x32_bf16(false, alo, false, bh, (short)0, accE[t], false, false);
            v16bf ch = *(const v16bf*)(pWh + g);
            v16bf cl = *(const v16bf*)(pWl + g);
            accW[t] = __builtin_amdgcn_wmma_f32_16x16x32_bf16(false, ahi, false, ch, (short)0, accW[t], false, false);
            accW[t] = __builtin_amdgcn_wmma_f32_16x16x32_bf16(false, ahi, false, cl, (short)0, accW[t], false, false);
            accW[t] = __builtin_amdgcn_wmma_f32_16x16x32_bf16(false, alo, false, ch, (short)0, accW[t], false, false);
        }
    }

    // epilogue: C/D layout -> VGPR r: lanes0-15 = (M=r, N=lane),
    //                                  lanes16-31 = (M=r+8, N=lane-16)
#pragma unroll
    for (int t = 0; t < 4; ++t) {
        int col = colb + t * 16;
        float beV = be[col];
        float bwV = bw[col];
#pragma unroll
        for (int r = 0; r < 8; ++r) {
            int token = m0 + r + laneHi * 8;
            size_t i0 = (size_t)token * DD + col;
            float xv = x[i0];
            // a = 1 - sigmoid(z) = sigmoid(-z) = 1/(1+exp(z))
            float aV = 1.0f / (1.0f + __expf(accE[t][r] + beV));
            float wV = 1.0f / (1.0f + __expf(-(accW[t][r] + bwV)));
            a_buf[i0] = aV;
            c_buf[i0] = wV * xv;
        }
    }
}

// ---------------------------------------------------------------------------
// Kernels 2-4: chunked parallel linear-recurrence scan.
//   s_l = a_l * s_{l-1} + c_l  composes as (A,B) -> (a*A, a*B + c)
// ---------------------------------------------------------------------------
__global__ void __launch_bounds__(256)
scan_partial_kernel(const float* __restrict__ a_buf, const float* __restrict__ c_buf,
                    float* __restrict__ A_part, float* __restrict__ B_part) {
    int d = blockIdx.x * blockDim.x + threadIdx.x;  // coalesced over d
    int j = blockIdx.y;
    int b = blockIdx.z;
    int l0 = j * CHUNK;
    float A = 1.0f, Bc = 0.0f;
    for (int l = 0; l < CHUNK; ++l) {
        size_t idx = ((size_t)(b * LL + l0 + l)) * DD + d;
        float a = a_buf[idx], c = c_buf[idx];
        Bc = a * Bc + c;
        A *= a;
    }
    int ch = b * DD + d;
    A_part[(size_t)j * BD + ch] = A;
    B_part[(size_t)j * BD + ch] = Bc;
}

__global__ void __launch_bounds__(256)
scan_prefix_kernel(const float* __restrict__ state,
                   const float* __restrict__ A_part, const float* __restrict__ B_part,
                   float* __restrict__ Sin) {
    int ch = blockIdx.x * blockDim.x + threadIdx.x;  // ch = b*D + d
    float s = state[ch];
#pragma unroll
    for (int j = 0; j < NCHUNK; ++j) {
        Sin[(size_t)j * BD + ch] = s;
        s = A_part[(size_t)j * BD + ch] * s + B_part[(size_t)j * BD + ch];
    }
}

__global__ void __launch_bounds__(256)
scan_final_kernel(const float* __restrict__ a_buf, const float* __restrict__ c_buf,
                  const float* __restrict__ Sin, float* __restrict__ out) {
    int d = blockIdx.x * blockDim.x + threadIdx.x;
    int j = blockIdx.y;
    int b = blockIdx.z;
    float s = Sin[(size_t)j * BD + b * DD + d];
    int l0 = j * CHUNK;
    for (int l = 0; l < CHUNK; ++l) {
        size_t idx = ((size_t)(b * LL + l0 + l)) * DD + d;
        s = a_buf[idx] * s + c_buf[idx];
        out[idx] = s;
    }
}

// ---------------------------------------------------------------------------
extern "C" void kernel_launch(void* const* d_in, const int* in_sizes, int n_in,
                              void* d_out, int out_size, void* d_ws, size_t ws_size,
                              hipStream_t stream) {
    const float* x     = (const float*)d_in[0];   // [B,L,D]
    const float* state = (const float*)d_in[1];   // [B,D]
    const float* We    = (const float*)d_in[2];   // [D,D]
    const float* be    = (const float*)d_in[3];   // [D]
    const float* Ww    = (const float*)d_in[4];   // [D,D]
    const float* bw    = (const float*)d_in[5];   // [D]
    float* out = (float*)d_out;

    // workspace layout
    float*  a_buf  = (float*)d_ws;                         // 16M f32 (64 MB)
    float*  c_buf  = a_buf + (size_t)BL * DD;              // 16M f32 (64 MB)
    __bf16* pack   = (__bf16*)(c_buf + (size_t)BL * DD);   // 4M bf16 (8 MB)
    float*  A_part = (float*)(pack + 4 * (size_t)PACK_PART);
    float*  B_part = A_part + (size_t)NCHUNK * BD;
    float*  Sin    = B_part + (size_t)NCHUNK * BD;

    pack_w_kernel<<<512, 256, 0, stream>>>(We, Ww, pack);

    gates_gemm_kernel<<<dim3(BL / 16, DD / 512), 256, 0, stream>>>(
        x, be, bw, pack, a_buf, c_buf);

    scan_partial_kernel<<<dim3(DD / 256, NCHUNK, BB), 256, 0, stream>>>(
        a_buf, c_buf, A_part, B_part);

    scan_prefix_kernel<<<BD / 256, 256, 0, stream>>>(state, A_part, B_part, Sin);

    scan_final_kernel<<<dim3(DD / 256, NCHUNK, BB), 256, 0, stream>>>(
        a_buf, c_buf, Sin, out);
}
